// Cross_VQEmbeddingEMA_87668872446335
// MI455X (gfx1250) — matrix operand order repaired and checked by reading, not compiled, not run on hardware
//
#include <hip/hip_runtime.h>
#include <hip/hip_bf16.h>

typedef __attribute__((ext_vector_type(16))) _Float16 v16h;
typedef __attribute__((ext_vector_type(8)))  _Float16 v8h;
typedef __attribute__((ext_vector_type(8)))  float    v8f;
typedef __attribute__((ext_vector_type(4)))  unsigned int v4u;
typedef __attribute__((ext_vector_type(8)))  int v8i;
typedef __attribute__((ext_vector_type(4)))  int v4i;

#define BB   64
#define TT   256
#define DD   512
#define MM   1024
#define NN   (BB*TT)          // 16384 rows per modality
#define CC_  0.25f

// ---------------------------------------------------------------------------
// Tensor DMA descriptor builders (CDNA5 D#, 2D tile, ISA 08_async_tensor §8)
// ---------------------------------------------------------------------------
__device__ __forceinline__ v4u tdm_g0(unsigned lds_addr, unsigned long long gaddr)
{
    v4u g;
    g[0] = 1u;                                            // count=1, no gather
    g[1] = lds_addr;                                      // LDS byte address
    g[2] = (unsigned)(gaddr & 0xFFFFFFFFu);               // global_addr[31:0]
    g[3] = (unsigned)((gaddr >> 32) & 0x1FFFFFFu)         // global_addr[56:32]
         | (2u << 30);                                    // type=2 ("image")
    return g;
}

__device__ __forceinline__ v8i tdm_g1_2d(unsigned tensor_d0, unsigned tensor_d1,
                                         unsigned tile_d0, unsigned tile_d1,
                                         unsigned stride0)
{
    v8i g;
    g[0] = (int)(1u << 16);                               // data_size=1 -> 2 bytes
    g[1] = (int)((tensor_d0 & 0xFFFFu) << 16);            // tensor_dim0[15:0]
    g[2] = (int)(((tensor_d0 >> 16) & 0xFFFFu)            // tensor_dim0[31:16]
         | ((tensor_d1 & 0xFFFFu) << 16));                // tensor_dim1[15:0]
    g[3] = (int)(((tensor_d1 >> 16) & 0xFFFFu)            // tensor_dim1[31:16]
         | ((tile_d0 & 0xFFFFu) << 16));                  // tile_dim0
    g[4] = (int)(tile_d1 & 0xFFFFu);                      // tile_dim1 (tile_dim2=0)
    g[5] = (int)stride0;                                  // dim0_stride[31:0]
    g[6] = 0;                                             // dim0_stride[47:32]=0
    g[7] = 0;
    return g;
}

// ---------------------------------------------------------------------------
// 1) fp32 -> f16 conversion + row squared-norms (rows of length 512)
// ---------------------------------------------------------------------------
__global__ __launch_bounds__(128) void prep_rows(const float* __restrict__ x,
                                                 _Float16* __restrict__ xh,
                                                 float* __restrict__ xn)
{
    const int row = blockIdx.x, tid = threadIdx.x;
    const float* xr = x + (size_t)row * DD;
    _Float16*    hr = xh + (size_t)row * DD;
    float s = 0.f;
#pragma unroll
    for (int j = 0; j < 4; ++j) {
        int d = tid + j * 128;
        float v = xr[d];
        hr[d] = (_Float16)v;
        s += v * v;
    }
    __shared__ float red[128];
    red[tid] = s; __syncthreads();
    for (int o = 64; o > 0; o >>= 1) { if (tid < o) red[tid] += red[tid + o]; __syncthreads(); }
    if (tid == 0) xn[row] = red[0];
}

// ---------------------------------------------------------------------------
// 2) dist = ||x||^2 + ||e||^2 - 2 * x . e    (N x M), f16 WMMA, f32 accum
//    block tile 128x128, 256 threads = 8 waves, wave w: rows [w*16,w*16+16)
//    A/B tiles staged to LDS by the Tensor Data Mover (wave 0 issues DMA).
// ---------------------------------------------------------------------------
__global__ __launch_bounds__(256) void dist_gemm(const _Float16* __restrict__ xh,
                                                 const _Float16* __restrict__ eh,
                                                 const float* __restrict__ xn,
                                                 const float* __restrict__ en,
                                                 float* __restrict__ dist)
{
    __shared__ _Float16 sA[128 * 32];
    __shared__ _Float16 sB[128 * 32];

    const int tid  = threadIdx.x;
    const int wid  = tid >> 5;
    const int lane = tid & 31;
    const int rowBase = blockIdx.y * 128;
    const int colBase = blockIdx.x * 128;

    v8f acc[8];
#pragma unroll
    for (int c = 0; c < 8; ++c) acc[c] = (v8f){0.f,0.f,0.f,0.f,0.f,0.f,0.f,0.f};

    const unsigned ldsA = (unsigned)(unsigned long long)(uintptr_t)(void*)sA;
    const unsigned ldsB = (unsigned)(unsigned long long)(uintptr_t)(void*)sB;
    const v4i z4 = (v4i){0, 0, 0, 0};
    const v8i z8 = (v8i){0, 0, 0, 0, 0, 0, 0, 0};

    for (int kk = 0; kk < DD; kk += 32) {
        // --- TDM: DMA the 128x32-half A and B tiles (row stride 512) into LDS
        if (wid == 0) {
            unsigned long long ga =
                (unsigned long long)(uintptr_t)(xh + (size_t)rowBase * DD + kk);
            unsigned long long gb =
                (unsigned long long)(uintptr_t)(eh + (size_t)colBase * DD + kk);
            v8i g1 = tdm_g1_2d(/*tensor_d0=*/DD - kk, /*tensor_d1=*/128,
                               /*tile_d0=*/32, /*tile_d1=*/128, /*stride0=*/DD);
            __builtin_amdgcn_tensor_load_to_lds(tdm_g0(ldsA, ga), g1, z4, z4, z8, 0);
            __builtin_amdgcn_tensor_load_to_lds(tdm_g0(ldsB, gb), g1, z4, z4, z8, 0);
            __builtin_amdgcn_s_wait_tensorcnt(0);
        }
        __syncthreads();

        // A fragment: 16x32 f16.  ISA layout: lanes 0-15 row=lane, K {0..7,16..23};
        // lanes 16-31 row=lane-16, K {8..15,24..31}.
        const int ar = wid * 16 + (lane & 15);
        const int ak = (lane < 16) ? 0 : 8;
        v8h alo = *(const v8h*)(&sA[ar * 32 + ak]);
        v8h ahi = *(const v8h*)(&sA[ar * 32 + ak + 16]);
        v16h afrag = __builtin_shufflevector(alo, ahi,
            0,1,2,3,4,5,6,7,8,9,10,11,12,13,14,15);

        // Preload all 8 B fragments so the ds_loads pipeline, then run the
        // 8 WMMAs back-to-back (shared A, distinct accumulators: no hazards).
        v16h bfr[8];
#pragma unroll
        for (int c = 0; c < 8; ++c) {
            const int bc = c * 16 + (lane & 15);
            const int bk = (lane < 16) ? 0 : 8;
            v8h blo = *(const v8h*)(&sB[bc * 32 + bk]);
            v8h bhi = *(const v8h*)(&sB[bc * 32 + bk + 16]);
            bfr[c] = __builtin_shufflevector(blo, bhi,
                0,1,2,3,4,5,6,7,8,9,10,11,12,13,14,15);
        }
#pragma unroll
        for (int c = 0; c < 8; ++c) {
            acc[c] = __builtin_amdgcn_wmma_f32_16x16x32_f16(
                false, afrag, false, bfr[c], (short)0, acc[c], false, false);
        }
        __syncthreads();
    }

    // Epilogue: C/D layout -> VGPR r holds M=r (lanes 0-15) / M=r+8 (lanes 16-31)
    const int rsub  = (lane < 16) ? 0 : 8;
    const int colIn = lane & 15;
    float xnv[8];
#pragma unroll
    for (int r = 0; r < 8; ++r) xnv[r] = xn[rowBase + wid * 16 + rsub + r];
#pragma unroll
    for (int c = 0; c < 8; ++c) {
        const int col = colBase + c * 16 + colIn;
        const float env = en[col];
#pragma unroll
        for (int r = 0; r < 8; ++r) {
            const int row = rowBase + wid * 16 + rsub + r;
            dist[(size_t)row * MM + col] = xnv[r] + env - 2.0f * acc[c][r];
        }
    }
}

// ---------------------------------------------------------------------------
// 3) per-row argmin (smallest index on ties) + softmax stats for -sqrt(dist)
//    one wave per row; 8 rows per block
// ---------------------------------------------------------------------------
__global__ __launch_bounds__(256) void row_reduce(const float* __restrict__ dist,
                                                  int* __restrict__ idx,
                                                  float* __restrict__ smin,
                                                  float* __restrict__ zsum)
{
    const int wid = threadIdx.x >> 5, lane = threadIdx.x & 31;
    const int row = blockIdx.x * 8 + wid;
    const float* dr = dist + (size_t)row * MM;

    float d[32];
    float mv = 3.4e38f; int mi = 0;
#pragma unroll
    for (int i = 0; i < 32; ++i) {
        const int m = lane + i * 32;
        const float v = dr[m];
        d[i] = v;
        if (v < mv) { mv = v; mi = m; }
    }
#pragma unroll
    for (int o = 16; o > 0; o >>= 1) {
        float ov = __shfl_xor(mv, o, 32);
        int   oi = __shfl_xor(mi, o, 32);
        if (ov < mv || (ov == mv && oi < mi)) { mv = ov; mi = oi; }
    }
    const float s = sqrtf(fmaxf(mv, 0.f));
    float z = 0.f;
#pragma unroll
    for (int i = 0; i < 32; ++i) z += expf(s - sqrtf(fmaxf(d[i], 0.f)));
#pragma unroll
    for (int o = 16; o > 0; o >>= 1) z += __shfl_xor(z, o, 32);
    if (lane == 0) { idx[row] = mi; smin[row] = s; zsum[row] = z; }
}

// ---------------------------------------------------------------------------
// 4) pH[b][m] = mean over t of softmax probs
// ---------------------------------------------------------------------------
__global__ __launch_bounds__(256) void ph_mean(const float* __restrict__ dist,
                                               const float* __restrict__ smin,
                                               const float* __restrict__ zs,
                                               float* __restrict__ pH)
{
    const int b = blockIdx.x, tid = threadIdx.x;
    float acc[4] = {0.f, 0.f, 0.f, 0.f};
    for (int t = 0; t < TT; ++t) {
        const int row = b * TT + t;
        const float s  = smin[row];
        const float zi = 1.0f / zs[row];
#pragma unroll
        for (int j = 0; j < 4; ++j) {
            const int m = tid + j * 256;
            acc[j] += expf(s - sqrtf(fmaxf(dist[(size_t)row * MM + m], 0.f))) * zi;
        }
    }
#pragma unroll
    for (int j = 0; j < 4; ++j)
        pH[(size_t)b * MM + tid + j * 256] = acc[j] * (1.0f / (float)TT);
}

// ---------------------------------------------------------------------------
// 5) consistency loss: sum |ph - pH_b| (accumulated into accum[slot])
// ---------------------------------------------------------------------------
__global__ __launch_bounds__(256) void consistency(const float* __restrict__ dist,
                                                   const float* __restrict__ smin,
                                                   const float* __restrict__ zs,
                                                   const float* __restrict__ pH,
                                                   float* __restrict__ accum, int slot)
{
    const int row = blockIdx.x, b = row >> 8, tid = threadIdx.x;
    const float s  = smin[row];
    const float zi = 1.0f / zs[row];
    float sum = 0.f;
#pragma unroll
    for (int j = 0; j < 4; ++j) {
        const int m = tid + j * 256;
        const float ph = expf(s - sqrtf(fmaxf(dist[(size_t)row * MM + m], 0.f))) * zi;
        sum += fabsf(ph - pH[(size_t)b * MM + m]);
    }
    __shared__ float red[256];
    red[tid] = sum; __syncthreads();
    for (int o = 128; o > 0; o >>= 1) { if (tid < o) red[tid] += red[tid + o]; __syncthreads(); }
    if (tid == 0) atomicAdd(&accum[slot], red[0]);
}

// ---------------------------------------------------------------------------
// 6) Scode (64x64) + cmcm loss, one block
// ---------------------------------------------------------------------------
__global__ __launch_bounds__(256) void cmcm_kernel(const float* __restrict__ aPH,
                                                   const float* __restrict__ vPH,
                                                   const int* __restrict__ epoch,
                                                   float* __restrict__ out_cmcm)
{
    __shared__ float S[64 * 64];
    __shared__ float red[256];
    __shared__ float lsum[64];
    const int tid = threadIdx.x;
    for (int e = tid; e < 4096; e += 256) {
        const int i = e >> 6, j = e & 63;
        float s = 0.f;
        for (int m = 0; m < MM; ++m) {
            s += aPH[(size_t)i * MM + m] * logf(vPH[(size_t)j * MM + m] + 1e-10f)
               + vPH[(size_t)i * MM + m] * logf(aPH[(size_t)j * MM + m] + 1e-10f);
        }
        S[e] = s;
    }
    __syncthreads();
    float mx = -3.4e38f;
    for (int e = tid; e < 4096; e += 256) mx = fmaxf(mx, -S[e]);
    red[tid] = mx; __syncthreads();
    for (int o = 128; o > 0; o >>= 1) { if (tid < o) red[tid] = fmaxf(red[tid], red[tid + o]); __syncthreads(); }
    const float mneg = red[0];
    __syncthreads();
    if (tid < 64) {
        float rs = 0.f;
        for (int j = 0; j < 64; ++j) rs += expf(S[tid * 64 + j] + mneg);
        const float diag = expf(S[tid * 64 + tid] + mneg);
        lsum[tid] = logf(diag / (rs + 1e-5f));
    }
    __syncthreads();
    if (tid == 0) {
        float tot = 0.f;
        for (int i = 0; i < 64; ++i) tot += lsum[i];
        const float L = -tot / 64.f;
        *out_cmcm = (epoch[0] < 10) ? 0.f : 0.5f * L;
    }
}

// ---------------------------------------------------------------------------
// 7) gather quantized vectors (straight-through forward value) + MSE partials
// ---------------------------------------------------------------------------
__global__ __launch_bounds__(256) void quant_commit(const float* __restrict__ a,
                                                    const float* __restrict__ v,
                                                    const float* __restrict__ emb,
                                                    const int* __restrict__ aidx,
                                                    const int* __restrict__ vidx,
                                                    float* __restrict__ aout,
                                                    float* __restrict__ vout,
                                                    float* __restrict__ accum)
{
    const int n = blockIdx.x, tid = threadIdx.x;
    const int ai = aidx[n], vi = vidx[n];
    float s0 = 0.f, s1 = 0.f, s2 = 0.f, s3 = 0.f;
#pragma unroll
    for (int j = 0; j < 2; ++j) {
        const int d = tid + j * 256;
        const float av = a[(size_t)n * DD + d], vv = v[(size_t)n * DD + d];
        const float aq = emb[(size_t)ai * DD + d], vq = emb[(size_t)vi * DD + d];
        aout[(size_t)n * DD + d] = aq;
        vout[(size_t)n * DD + d] = vq;
        const float d0 = av - aq, d1 = av - vq, d2 = vv - vq, d3 = vv - aq;
        s0 += d0 * d0; s1 += d1 * d1; s2 += d2 * d2; s3 += d3 * d3;
    }
#pragma unroll
    for (int o = 16; o > 0; o >>= 1) {
        s0 += __shfl_xor(s0, o, 32); s1 += __shfl_xor(s1, o, 32);
        s2 += __shfl_xor(s2, o, 32); s3 += __shfl_xor(s3, o, 32);
    }
    __shared__ float red[8][4];
    const int wid = tid >> 5, lane = tid & 31;
    if (lane == 0) { red[wid][0] = s0; red[wid][1] = s1; red[wid][2] = s2; red[wid][3] = s3; }
    __syncthreads();
    if (tid < 4) {
        float t = 0.f;
        for (int w = 0; w < 8; ++w) t += red[w][tid];
        atomicAdd(&accum[tid], t);
    }
}

// ---------------------------------------------------------------------------
// 8) per-clip mode (smallest index on ties) + equal count
// ---------------------------------------------------------------------------
__global__ __launch_bounds__(256) void mode_equal(const int* __restrict__ aidx,
                                                  const int* __restrict__ vidx,
                                                  float* __restrict__ accum)
{
    __shared__ int cnt[MM];
    __shared__ int bestc[256];
    __shared__ int besti[256];
    __shared__ int modes_s[2];
    const int b = blockIdx.x, tid = threadIdx.x;
    for (int pass = 0; pass < 2; ++pass) {
        const int* idx = pass ? vidx : aidx;
        for (int j = tid; j < MM; j += 256) cnt[j] = 0;
        __syncthreads();
        atomicAdd(&cnt[idx[b * TT + tid]], 1);
        __syncthreads();
        int bc = -1, bi = 0;
#pragma unroll
        for (int j = 0; j < 4; ++j) {
            const int m = tid * 4 + j;
            const int c = cnt[m];
            if (c > bc) { bc = c; bi = m; }
        }
        bestc[tid] = bc; besti[tid] = bi;
        __syncthreads();
        for (int o = 128; o > 0; o >>= 1) {
            if (tid < o) {
                const int c2 = bestc[tid + o], i2 = besti[tid + o];
                if (c2 > bestc[tid] || (c2 == bestc[tid] && i2 < besti[tid])) {
                    bestc[tid] = c2; besti[tid] = i2;
                }
            }
            __syncthreads();
        }
        if (tid == 0) modes_s[pass] = besti[0];
        __syncthreads();
    }
    if (tid == 0 && modes_s[0] == modes_s[1]) atomicAdd(&accum[6], 1.0f);
}

// ---------------------------------------------------------------------------
// 9) scalars
// ---------------------------------------------------------------------------
__global__ void init_accum(float* accum)
{
    if (threadIdx.x < 16) accum[threadIdx.x] = 0.f;
}

__global__ void finalize(const float* __restrict__ accum, float* __restrict__ out)
{
    if (threadIdx.x == 0 && blockIdx.x == 0) {
        const float inv = 1.0f / (float)((size_t)NN * DD);
        const float a_e = accum[0] * inv, av_e = accum[1] * inv;
        const float v_e = accum[2] * inv, va_e = accum[3] * inv;
        out[0] = CC_ * 2.0f * a_e + CC_ * av_e;   // a_loss
        out[1] = CC_ * 2.0f * v_e + CC_ * va_e;   // v_loss
        // out[2] = cmcm (written by cmcm_kernel)
        out[3] = accum[4] / (float)BB;            // a_consistency
        out[4] = accum[5] / (float)BB;            // v_consistency
        out[5] = accum[6];                        // equal_num
    }
}

// ---------------------------------------------------------------------------
extern "C" void kernel_launch(void* const* d_in, const int* in_sizes, int n_in,
                              void* d_out, int out_size, void* d_ws, size_t ws_size,
                              hipStream_t stream)
{
    const float* audio = (const float*)d_in[0];
    const float* video = (const float*)d_in[1];
    const float* emb   = (const float*)d_in[2];
    const int*   epoch = (const int*)d_in[3];
    float* out = (float*)d_out;
    const size_t BTD = (size_t)NN * DD;

    char* ws = (char*)d_ws;
    size_t off = 0;
    auto take = [&](size_t bytes) -> void* {
        void* p = ws + off;
        off = (off + bytes + 255) & ~(size_t)255;
        return p;
    };
    _Float16* a_h   = (_Float16*)take((size_t)NN * DD * 2);
    _Float16* v_h   = (_Float16*)take((size_t)NN * DD * 2);
    _Float16* e_h   = (_Float16*)take((size_t)MM * DD * 2);
    float*    a_nrm = (float*)take((size_t)NN * 4);
    float*    v_nrm = (float*)take((size_t)NN * 4);
    float*    e_nrm = (float*)take((size_t)MM * 4);
    float*    a_dst = (float*)take((size_t)NN * MM * 4);
    float*    v_dst = (float*)take((size_t)NN * MM * 4);
    int*      a_idx = (int*)take((size_t)NN * 4);
    int*      v_idx = (int*)take((size_t)NN * 4);
    float*    a_sm  = (float*)take((size_t)NN * 4);
    float*    a_z   = (float*)take((size_t)NN * 4);
    float*    v_sm  = (float*)take((size_t)NN * 4);
    float*    v_z   = (float*)take((size_t)NN * 4);
    float*    a_pH  = (float*)take((size_t)BB * MM * 4);
    float*    v_pH  = (float*)take((size_t)BB * MM * 4);
    float*    accum = (float*)take(64);

    init_accum<<<1, 64, 0, stream>>>(accum);

    prep_rows<<<NN, 128, 0, stream>>>(audio, a_h, a_nrm);
    prep_rows<<<NN, 128, 0, stream>>>(video, v_h, v_nrm);
    prep_rows<<<MM, 128, 0, stream>>>(emb,   e_h, e_nrm);

    dim3 g(MM / 128, NN / 128);
    dist_gemm<<<g, 256, 0, stream>>>(a_h, e_h, a_nrm, e_nrm, a_dst);
    dist_gemm<<<g, 256, 0, stream>>>(v_h, e_h, v_nrm, e_nrm, v_dst);

    row_reduce<<<NN / 8, 256, 0, stream>>>(a_dst, a_idx, a_sm, a_z);
    row_reduce<<<NN / 8, 256, 0, stream>>>(v_dst, v_idx, v_sm, v_z);

    ph_mean<<<BB, 256, 0, stream>>>(a_dst, a_sm, a_z, a_pH);
    ph_mean<<<BB, 256, 0, stream>>>(v_dst, v_sm, v_z, v_pH);

    consistency<<<NN, 256, 0, stream>>>(a_dst, a_sm, a_z, a_pH, accum, 4);
    consistency<<<NN, 256, 0, stream>>>(v_dst, v_sm, v_z, v_pH, accum, 5);

    cmcm_kernel<<<1, 256, 0, stream>>>(a_pH, v_pH, epoch, out + 2 * BTD + 2);

    quant_commit<<<NN, 256, 0, stream>>>(audio, video, emb, a_idx, v_idx,
                                         out, out + BTD, accum);

    mode_equal<<<BB, 256, 0, stream>>>(a_idx, v_idx, accum);

    finalize<<<1, 1, 0, stream>>>(accum, out + 2 * BTD);
}
